// SQConvT2d_85684597555512
// MI455X (gfx1250) — 8-bit matrix operand order repaired and checked by reading, compiled, not run on hardware
//
#include <hip/hip_runtime.h>
#include <math.h>

typedef __attribute__((ext_vector_type(8))) int v8i;

#define IC   256
#define OC   128
#define KW   3
#define BS   16
#define IH   64
#define IW   64
#define OHT  128
#define OWT  128
#define NCOL (OC * KW * KW)
#define M_TOT (BS * IH * IW)
#define MT   (M_TOT / 16)
#define NTI  (NCOL / 16)
#define KCN  (IC / 64)

__global__ void k_wmax(const float* __restrict__ w, float* __restrict__ wmax) {
    __shared__ float red[256];
    const int ic = blockIdx.x;
    const float* row = w + (size_t)ic * NCOL;
    float m = 0.f;
    for (int i = threadIdx.x; i < NCOL; i += 256) m = fmaxf(m, fabsf(row[i]));
    red[threadIdx.x] = m; __syncthreads();
    for (int s = 128; s > 0; s >>= 1) {
        if (threadIdx.x < s) red[threadIdx.x] = fmaxf(red[threadIdx.x], red[threadIdx.x + s]);
        __syncthreads();
    }
    if (threadIdx.x == 0) wmax[ic] = red[0];
}

__global__ void k_xmax(const float* __restrict__ x, float* __restrict__ xmax) {
    __shared__ float red[256];
    const int ic = blockIdx.x;
    float m = 0.f;
    for (int b = 0; b < BS; ++b) {
        const float* p = x + ((size_t)(b * IC + ic)) * (IH * IW);
        for (int i = threadIdx.x; i < IH * IW; i += 256) m = fmaxf(m, fabsf(p[i]));
    }
    red[threadIdx.x] = m; __syncthreads();
    for (int s = 128; s > 0; s >>= 1) {
        if (threadIdx.x < s) red[threadIdx.x] = fmaxf(red[threadIdx.x], red[threadIdx.x + s]);
        __syncthreads();
    }
    if (threadIdx.x == 0) xmax[ic] = red[0];
}

__global__ void k_scale(float* __restrict__ wsf) {
    __shared__ float rw[256], rx[256];
    const int c = threadIdx.x;
    const float wm = wsf[c];
    const float xm = wsf[256 + c];
    float s = sqrtf(xm) / sqrtf(wm);
    if (s == 0.f) s = 1.f;
    rw[c] = wm * s;
    rx[c] = xm / s;
    __syncthreads();
    for (int k = 128; k > 0; k >>= 1) {
        if (c < k) { rw[c] = fmaxf(rw[c], rw[c + k]); rx[c] = fmaxf(rx[c], rx[c + k]); }
        __syncthreads();
    }
    float sw = rw[0] * (1.f / 127.f); if (sw == 0.f) sw = 1.f;
    float sx = rx[0] * (1.f / 127.f); if (sx == 0.f) sx = 1.f;
    wsf[512 + c] = 1.f / (s * sx);
    wsf[768 + c] = s / sw;
    if (c == 0) wsf[1024] = sw * sx;
}

__global__ void k_qx(const float* __restrict__ x, const float* __restrict__ a_mul,
                     signed char* __restrict__ A) {
    const int idx = blockIdx.x * 256 + threadIdx.x;
    const int j  = idx & 31;
    const int l  = (idx >> 5) & 31;
    const int kc = (idx >> 10) & 3;
    const int mt = idx >> 12;
    const int d  = j >> 2;
    const int koff = ((l < 16) ? 0 : 8) + (d >> 1) * 16 + (d & 1) * 4 + (j & 3);
    const int k = kc * 64 + koff;
    const int m = mt * 16 + (l & 15);
    const int b = m >> 12, hw = m & 4095;
    float v = x[(((size_t)b * IC + k) << 12) + hw] * a_mul[k];
    v = rintf(v);
    v = fminf(fmaxf(v, -128.f), 127.f);
    A[idx] = (signed char)(int)v;
}

__global__ void k_qw(const float* __restrict__ w, const float* __restrict__ w_mul,
                     signed char* __restrict__ B) {
    const int idx = blockIdx.x * 256 + threadIdx.x;
    const int j  = idx & 31;
    const int l  = (idx >> 5) & 31;
    const int kc = (idx >> 10) & 3;
    const int nt = idx >> 12;
    const int d  = j >> 2;
    const int half = (l < 16) ? 0 : 16;
    const int koff = (half >> 1) + (d >> 1) * 16 + (d & 1) * 4 + (j & 3);
    const int k = kc * 64 + koff;
    const int r = nt * 16 + (l & 15);
    float v = w[(size_t)k * NCOL + r] * w_mul[k];
    v = rintf(v);
    v = fminf(fmaxf(v, -128.f), 127.f);
    B[idx] = (signed char)(int)v;
}

__global__ void k_bias(const float* __restrict__ bias, float* __restrict__ out) {
    const int idx = blockIdx.x * 256 + threadIdx.x;
    out[idx] = bias[(idx >> 14) & (OC - 1)];
}

__global__ void __launch_bounds__(256) k_gemm(const signed char* __restrict__ A8,
                                              const signed char* __restrict__ B8,
                                              const float* __restrict__ deqp,
                                              float* __restrict__ out) {
    const int lane = threadIdx.x & 31;
    const int wave = (blockIdx.x * 256 + threadIdx.x) >> 5;
    const int mt  = wave / 18;
    const int ng  = wave - mt * 18;
    const int nt0 = ng * 4;
    const float deq = deqp[0];

    v8i a[4];
#pragma unroll
    for (int kc = 0; kc < 4; ++kc)
        a[kc] = *(const v8i*)(A8 + (((size_t)(mt * 4 + kc) * 32 + lane) << 5));

    v8i acc[4];
#pragma unroll
    for (int t = 0; t < 4; ++t) {
        v8i c = {};
#pragma unroll
        for (int kc = 0; kc < 4; ++kc) {
            v8i b = *(const v8i*)(B8 + ((((size_t)(nt0 + t) * 4 + kc) * 32 + lane) << 5));
            c = __builtin_amdgcn_wmma_i32_16x16x64_iu8(true, a[kc], true, b, c, false, false);
        }
        acc[t] = c;
    }

    const int n_lo  = lane & 15;
    const int m_hi  = (lane >> 4) << 3;
#pragma unroll
    for (int t = 0; t < 4; ++t) {
        const int col = (nt0 + t) * 16 + n_lo;
        const int oc  = col / 9;
        const int rem = col - oc * 9;
        const int ki  = rem / 3;
        const int kj  = rem - ki * 3;
#pragma unroll
        for (int r = 0; r < 8; ++r) {
            const float v = (float)acc[t][r] * deq;
            const int m  = mt * 16 + r + m_hi;
            const int b  = m >> 12, hw = m & 4095;
            const int ih = hw >> 6, iw = hw & 63;
            const int oh = 2 * ih + ki - 1;
            const int ow = 2 * iw + kj - 1;
            if ((unsigned)oh < (unsigned)OHT && (unsigned)ow < (unsigned)OWT) {
                size_t off = (((size_t)b * OC + oc) * OHT + oh) * OWT + ow;
                unsafeAtomicAdd(out + off, v);
            }
        }
    }
}

extern "C" void kernel_launch(void* const* d_in, const int* in_sizes, int n_in,
                              void* d_out, int out_size, void* d_ws, size_t ws_size,
                              hipStream_t stream) {
    const float* x    = (const float*)d_in[0];
    const float* w    = (const float*)d_in[1];
    const float* bias = (const float*)d_in[2];
    float* out = (float*)d_out;
    float* wsf = (float*)d_ws;

    signed char* A8 = (signed char*)d_ws + 8192;
    signed char* B8 = A8 + (size_t)MT * KCN * 32 * 32;

    k_wmax<<<IC, 256, 0, stream>>>(w, wsf);
    k_xmax<<<IC, 256, 0, stream>>>(x, wsf + 256);
    k_scale<<<1, 256, 0, stream>>>(wsf);
    k_qx<<<(MT * KCN * 1024) / 256, 256, 0, stream>>>(x, wsf + 512, A8);
    k_qw<<<(NTI * KCN * 1024) / 256, 256, 0, stream>>>(w, wsf + 768, B8);
    k_bias<<<(BS * OC * OHT * OWT) / 256, 256, 0, stream>>>(bias, out);
    k_gemm<<<(MT * (NTI / 4)) / 8, 256, 0, stream>>>(A8, B8, wsf + 1024, out);
}
